// GATV2Layer_67448166416668
// MI455X (gfx1250) — compile-verified, matrix operand order checked by reading, not run on hardware
//
#include <hip/hip_runtime.h>

typedef __attribute__((ext_vector_type(16))) __bf16 v16bf;
typedef __attribute__((ext_vector_type(8)))  float  v8f;

#define BS     2
#define NN     512
#define DIN    256
#define HEADS  8
#define DOUT   64
#define HD     (HEADS * DOUT)      /* 512 */
#define NEG_SLOPE 0.2f
#define MASK_VAL  -1000000000.0f

// ---- gfx1250 async global->LDS path (guarded; falls back to float4 copies) --
#if defined(__has_builtin)
#if __has_builtin(__builtin_amdgcn_global_load_async_to_lds_b128) && \
    __has_builtin(__builtin_amdgcn_s_wait_asynccnt)
#define USE_ASYNC 1
#endif
#endif
#ifndef USE_ASYNC
#define USE_ASYNC 0
#endif

// Builtin parameter types per hipcc diagnostic:
//   (v4i addrspace(1)*, v4i addrspace(3)*, imm int, imm int)
typedef int v4i __attribute__((vector_size(4 * sizeof(int))));
typedef __attribute__((address_space(1))) v4i gv4i;
typedef __attribute__((address_space(3))) v4i lv4i;

__device__ __forceinline__ void copy16_g2l(const void* g, void* l) {
#if USE_ASYNC
    __builtin_amdgcn_global_load_async_to_lds_b128(
        (gv4i*)(uintptr_t)g, (lv4i*)(uintptr_t)l, 0, 0);
#else
    *(float4*)l = *(const float4*)g;
#endif
}
__device__ __forceinline__ void async_wait0() {
#if USE_ASYNC
    __builtin_amdgcn_s_wait_asynccnt(0);
#endif
}

// ---------------------------------------------------------------------------
// Kernel 1: fused projection GEMMs.  O[z] = X(1024x256) @ W_z(256x512), z=0..2
// bf16 operands, f32 accumulate via v_wmma_f32_16x16x32_bf16.
// For z==0 (src) additionally emits a bf16 copy transposed to [b][h][d][j]
// so the attention kernel's V staging becomes a pure async byte copy.
// ---------------------------------------------------------------------------
__global__ __launch_bounds__(128) void proj_gemm_kernel(
    const float* __restrict__ X, const float* __restrict__ Wsrc,
    const float* __restrict__ Wtgt, const float* __restrict__ Wskp,
    float* __restrict__ wsOut)
{
    const int which = blockIdx.z;
    const float* W = (which == 0) ? Wsrc : (which == 1) ? Wtgt : Wskp;
    float* O = wsOut + (size_t)which * (BS * NN) * HD;
    __bf16* srcT = (__bf16*)(wsOut + (size_t)3 * (BS * NN) * HD);

    const int m0 = blockIdx.y * 16;
    const int n0 = blockIdx.x * 64;
    const int t    = threadIdx.x;
    const int wave = t >> 5;
    const int lane = t & 31;
    const int lm   = lane & 15;      // M (A) / N (B) index within tile
    const int hi   = lane >> 4;      // K-group selector per 16-bit WMMA layout

    __shared__ __bf16 Alds[16][DIN + 8];   // [m][k]
    __shared__ __bf16 Blds[64][DIN + 8];   // [n][k]  (W staged transposed)

    #pragma unroll 4
    for (int e = 0; e < 32; ++e) {
        int idx = e * 128 + t;
        int r = idx >> 8, k = idx & 255;
        Alds[r][k] = (__bf16)X[(size_t)(m0 + r) * DIN + k];
    }
    #pragma unroll 4
    for (int e = 0; e < 128; ++e) {
        int idx = e * 128 + t;
        int k = idx >> 6, n = idx & 63;
        Blds[n][k] = (__bf16)W[(size_t)k * HD + n0 + n];
    }
    __syncthreads();

    v8f acc = {};
    const __bf16* Arow = &Alds[lm][0];
    const __bf16* Brow = &Blds[wave * 16 + lm][0];

    #pragma unroll
    for (int kk = 0; kk < 8; ++kk) {        // K = 8 * 32
        const int kb = kk * 32;
        v16bf a, bfrag;
        #pragma unroll
        for (int q = 0; q < 4; ++q) {
            int k0 = kb + hi * 8 + 2 * q;
            a[2*q]         = Arow[k0];
            a[2*q+1]       = Arow[k0 + 1];
            a[2*(q+4)]     = Arow[k0 + 16];
            a[2*(q+4)+1]   = Arow[k0 + 17];
            bfrag[2*q]       = Brow[k0];
            bfrag[2*q+1]     = Brow[k0 + 1];
            bfrag[2*(q+4)]   = Brow[k0 + 16];
            bfrag[2*(q+4)+1] = Brow[k0 + 17];
        }
        acc = __builtin_amdgcn_wmma_f32_16x16x32_bf16(
                  false, a, false, bfrag, (short)0, acc, false, false);
    }

    const int col = n0 + wave * 16 + lm;    // = h*64 + d
    #pragma unroll
    for (int r = 0; r < 8; ++r) {
        const int row = m0 + r + 8 * hi;    // = b*512 + j
        O[(size_t)row * HD + col] = acc[r];
        if (which == 0) {
            const int bb = row >> 9, j = row & (NN - 1);
            const int hh = col >> 6, dd = col & 63;
            srcT[((size_t)((bb * HEADS + hh) * DOUT + dd)) * NN + j] = (__bf16)acc[r];
        }
    }
}

// ---------------------------------------------------------------------------
// Kernel 2: fused GATv2 attention per (b, h, 16-row tile).
// ---------------------------------------------------------------------------
__global__ __launch_bounds__(128) void gat_attn_kernel(
    const float* __restrict__ ws, const int* __restrict__ adj,
    const float* __restrict__ scoring, const float* __restrict__ bias,
    float* __restrict__ out)
{
    const float*  srcP = ws;                               // (1024, 512) f32
    const float*  tgtP = ws + (size_t)(BS * NN) * HD;
    const float*  skpP = ws + (size_t)2 * (BS * NN) * HD;
    const __bf16* srcT = (const __bf16*)(ws + (size_t)3 * (BS * NN) * HD);

    const int i0 = blockIdx.x * 16;
    const int h  = blockIdx.y;
    const int b  = blockIdx.z;
    const int t    = threadIdx.x;
    const int wave = t >> 5;
    const int lane = t & 31;
    const int lm   = lane & 15;
    const int hi   = lane >> 4;

    __shared__ float  sS[16][DOUT];          // src row tile (f32, async copy)
    __shared__ float  sA[DOUT];              // scoring vector for head h
    __shared__ float  sT[32][DOUT + 4];      // tgt chunk (272B rows, 16B multiple)
    __shared__ float  sScore[16][NN + 4];    // full score panel (f32)
    __shared__ __bf16 sP[16][NN + 8];        // softmax probs (bf16 A-fragments)
    __shared__ __bf16 sV[2][DOUT][40];       // V chunk [d][j], 80B rows, dbl-buffered

    const __bf16* vBase = srcT + (size_t)((b * HEADS + h) * DOUT) * NN;  // [d][j]

    // Stage src row tile (pure f32 copy: 16 rows x 16 segs of 16B)
    #pragma unroll
    for (int e = 0; e < 2; ++e) {
        int seg = e * 128 + t;
        int i = seg >> 4, s = seg & 15;
        copy16_g2l(srcP + (size_t)(b * NN + i0 + i) * HD + h * DOUT + s * 4,
                   &sS[i][s * 4]);
    }
    if (t < DOUT) sA[t] = scoring[h * DOUT + t];

    // ---- Phase 1: pairwise scoring, 16 chunks of 32 columns -----------------
    for (int jt = 0; jt < 16; ++jt) {
        const int j0 = jt * 32;
        __syncthreads();
        #pragma unroll
        for (int e = 0; e < 4; ++e) {        // 32 rows x 16 segs of 16B
            int seg = e * 128 + t;
            int j = seg >> 4, s = seg & 15;
            copy16_g2l(tgtP + (size_t)(b * NN + j0 + j) * HD + h * DOUT + s * 4,
                       &sT[j][s * 4]);
        }
        async_wait0();
        __syncthreads();
        #pragma unroll
        for (int e = 0; e < 4; ++e) {        // 512 pairs / 128 threads
            int pid = e * 128 + t;
            int i = pid >> 5, jl = pid & 31;
            float acc = 0.f;
            #pragma unroll 8
            for (int d = 0; d < DOUT; ++d) {
                float x = sS[i][d] + sT[jl][d];
                x = fmaxf(x, NEG_SLOPE * x);           // LeakyReLU (slope<1)
                acc = fmaf(sA[d], x, acc);
            }
            int av = adj[(size_t)b * NN * NN + (size_t)(i0 + i) * NN + (j0 + jl)];
            sScore[i][j0 + jl] = (av > 0) ? acc : (acc + MASK_VAL);
        }
    }
    __syncthreads();

    // ---- Phase 2: masked softmax, one wave handles 4 rows -------------------
    for (int rr = 0; rr < 4; ++rr) {
        const int i = wave * 4 + rr;
        float mx = -3.0e38f;
        for (int c = lane; c < NN; c += 32) mx = fmaxf(mx, sScore[i][c]);
        #pragma unroll
        for (int off = 16; off > 0; off >>= 1) mx = fmaxf(mx, __shfl_xor(mx, off, 32));
        float sum = 0.f;
        for (int c = lane; c < NN; c += 32) {
            float p = __expf(sScore[i][c] - mx);
            sScore[i][c] = p;
            sum += p;
        }
        #pragma unroll
        for (int off = 16; off > 0; off >>= 1) sum += __shfl_xor(sum, off, 32);
        const float inv = 1.0f / sum;
        for (int c = lane; c < NN; c += 32) sP[i][c] = (__bf16)(sScore[i][c] * inv);
    }
    __syncthreads();

    // ---- Phase 3: out(16x64) = P(16x512) @ V(512x64), double-buffered -------
    // stage V chunk jt into buffer buf: 64 rows x 4 segs of 16B (pure bf16 copy)
    auto stageV = [&](int jt, int buf) {
        #pragma unroll
        for (int e = 0; e < 2; ++e) {
            int seg = e * 128 + t;
            int d = seg >> 2, s = seg & 3;
            copy16_g2l(vBase + (size_t)d * NN + jt * 32 + s * 8,
                       &sV[buf][d][s * 8]);
        }
    };

    v8f acc = {};
    const int d0 = wave * 16;                // each wave owns a 16-wide d slab
    stageV(0, 0);
    for (int jt = 0; jt < 16; ++jt) {
        const int buf = jt & 1;
        stageV((jt + 1) & 15, buf ^ 1);      // last iter: harmless dummy issue
#if USE_ASYNC
        __builtin_amdgcn_s_wait_asynccnt(2); // this thread's chunk jt landed
#endif
        __syncthreads();
        v16bf a, v;
        const __bf16* Prow = &sP[lm][jt * 32];
        const __bf16* Vrow = &sV[buf][d0 + lm][0];
        #pragma unroll
        for (int q = 0; q < 4; ++q) {
            int k0 = hi * 8 + 2 * q;
            a[2*q]       = Prow[k0];
            a[2*q+1]     = Prow[k0 + 1];
            a[2*(q+4)]   = Prow[k0 + 16];
            a[2*(q+4)+1] = Prow[k0 + 17];
            v[2*q]       = Vrow[k0];
            v[2*q+1]     = Vrow[k0 + 1];
            v[2*(q+4)]   = Vrow[k0 + 16];
            v[2*(q+4)+1] = Vrow[k0 + 17];
        }
        acc = __builtin_amdgcn_wmma_f32_16x16x32_bf16(
                  false, a, false, v, (short)0, acc, false, false);
        __syncthreads();
    }

    // ---- Epilogue: skip + bias + ELU, write (b, i, h*64+d) ------------------
    #pragma unroll
    for (int r = 0; r < 8; ++r) {
        const int i  = r + 8 * hi;
        const int gd = h * DOUT + d0 + lm;
        const size_t gi = (size_t)(b * NN + i0 + i);
        float x = acc[r] + skpP[gi * HD + gd] + bias[gd];
        float y = (x > 0.f) ? x : (__expf(x) - 1.0f);
        out[gi * HD + gd] = y;
    }
}

// ---------------------------------------------------------------------------
extern "C" void kernel_launch(void* const* d_in, const int* in_sizes, int n_in,
                              void* d_out, int out_size, void* d_ws, size_t ws_size,
                              hipStream_t stream) {
    (void)in_sizes; (void)n_in; (void)out_size; (void)ws_size;
    const float* x    = (const float*)d_in[0];
    const int*   adj  = (const int*)d_in[1];
    const float* Wsrc = (const float*)d_in[2];
    const float* Wtgt = (const float*)d_in[3];
    const float* Wskp = (const float*)d_in[4];
    const float* scor = (const float*)d_in[5];
    const float* bia  = (const float*)d_in[6];

    float* ws  = (float*)d_ws;   // src|tgt|skip f32 (6MB) + srcT bf16 (1MB)
    float* out = (float*)d_out;

    proj_gemm_kernel<<<dim3(HD / 64, (BS * NN) / 16, 3), 128, 0, stream>>>(
        x, Wsrc, Wtgt, Wskp, ws);
    gat_attn_kernel<<<dim3(NN / 16, HEADS, BS), 128, 0, stream>>>(
        ws, adj, scor, bia, out);

    // Second tuple output: adj passed through (int32, same 4B width).
    (void)hipMemcpyAsync(out + (size_t)BS * NN * HD, adj,
                         (size_t)BS * NN * NN * sizeof(int),
                         hipMemcpyDeviceToDevice, stream);
}
